// MultiHeadAttention_84739704750531
// MI455X (gfx1250) — matrix operand order repaired and checked by reading, not compiled, not run on hardware
//
#include <hip/hip_runtime.h>
#include <stdint.h>

// ---------------------------------------------------------------------------
// MHA for MI455X (gfx1250). ~69 GFLOP vs ~34MB unique bytes -> matrix-core
// bound. All matmuls run on v_wmma_f32_16x16x32_bf16. Projection GEMMs stage
// weight panels into LDS via the Tensor Data Mover, DOUBLE-BUFFERED so the
// DMA of slab k+1 overlaps the 8 WMMAs of slab k:
//   wait tensorcnt(slab k) -> barrier -> issue slab k+1 (other buf) -> compute
// ---------------------------------------------------------------------------

typedef __bf16 bf16;
typedef __bf16 v16bf __attribute__((ext_vector_type(16)));
typedef __bf16 v8bf  __attribute__((ext_vector_type(8)));
typedef float  v8f   __attribute__((ext_vector_type(8)));
typedef unsigned u32x4 __attribute__((ext_vector_type(4)));
typedef unsigned u32x8 __attribute__((ext_vector_type(8)));

#define D_MODEL 1024
#define SEQ     2048
#define BATCH   2
#define HEADS   16
#define DK      64
#define SLAB    (64 * 32)     // bf16 elements per B slab (4 KiB)

static __device__ __forceinline__ v16bf combine8(v8bf lo, v8bf hi) {
  v16bf r;
#pragma unroll
  for (int i = 0; i < 8; ++i) { r[i] = lo[i]; r[i + 8] = hi[i]; }
  return r;
}

static __device__ __forceinline__ v8f wmma_bf16(v16bf a, v16bf b, v8f c) {
  return __builtin_amdgcn_wmma_f32_16x16x32_bf16(false, a, false, b,
                                                 (short)0, c, false, false);
}

// A-operand (16x32 bf16): lane%16 = row, K interleave half*8 / 16+half*8.
static __device__ __forceinline__ v16bf load_afrag(const bf16* row, int half) {
  return combine8(*(const v8bf*)(row + half * 8),
                  *(const v8bf*)(row + 16 + half * 8));
}
// B-operand (32x16 bf16): lane%16 = col, K interleave half*8 / 16+half*8.
static __device__ __forceinline__ v16bf load_bfrag(const bf16* col, int half) {
  return combine8(*(const v8bf*)(col + half * 8),
                  *(const v8bf*)(col + 16 + half * 8));
}

// ---------------------------------------------------------------------------
// TDM: DMA a 64(row) x 32(k) bf16 tile of a [1024][1024] bf16 matrix into
// LDS.  2-D tensor -> descriptor groups 2/3 omitted (D# per ISA §8.3/8.4).
// ---------------------------------------------------------------------------
static __device__ __forceinline__ void tdm_load_slab(const bf16* gsrc, unsigned lds_off) {
  uint64_t ga = (uint64_t)(uintptr_t)gsrc;
  u32x4 g0;
  g0[0] = 1u;                                            // count=1, user mode
  g0[1] = lds_off;                                       // lds_addr (bytes)
  g0[2] = (unsigned)(ga & 0xffffffffu);                  // global_addr[31:0]
  g0[3] = (unsigned)((ga >> 32) & 0x01ffffffu)           // global_addr[56:32]
        | (2u << 30);                                    // type = 2 ("image")
  u32x8 g1;
  g1[0] = (1u << 16);          // workgroup_mask=0, data_size=1 (2 bytes)
  g1[1] = (1024u << 16);       // tensor_dim0[15:0]=1024
  g1[2] = (1024u << 16);       // tensor_dim0 hi=0, tensor_dim1[15:0]=1024
  g1[3] = (32u << 16);         // tensor_dim1 hi=0, tile_dim0=32 (k)
  g1[4] = 64u;                 // tile_dim1=64 (rows), tile_dim2=0
  g1[5] = 1024u;               // tensor_dim0_stride=1024 elements
  g1[6] = 0u;
  g1[7] = 0u;
  asm volatile("tensor_load_to_lds %0, %1" :: "s"(g0), "s"(g1) : "memory");
}

// ---------------------------------------------------------------------------
// Kernel 0: cast x (f32) -> bf16.
// ---------------------------------------------------------------------------
__global__ void xcast_kernel(const float* __restrict__ x, bf16* __restrict__ xb) {
  int i = blockIdx.x * blockDim.x + threadIdx.x;
  xb[i] = (bf16)x[i];
}

// ---------------------------------------------------------------------------
// Kernel 1: cast+transpose the four f32 weights into bf16 [N][K].
// ---------------------------------------------------------------------------
__global__ void wtrans_kernel(const float* __restrict__ Wq, const float* __restrict__ Wk,
                              const float* __restrict__ Wv, const float* __restrict__ Wo,
                              bf16* __restrict__ dst) {
  const float* srcs[4] = {Wq, Wk, Wv, Wo};
  const float* W = srcs[blockIdx.y];
  bf16* out = dst + (size_t)blockIdx.y * D_MODEL * D_MODEL;
  int idx = blockIdx.x * blockDim.x + threadIdx.x;
  int n = idx >> 10, k = idx & (D_MODEL - 1);
  out[(size_t)n * D_MODEL + k] = (bf16)W[(size_t)k * D_MODEL + n];
}

// ---------------------------------------------------------------------------
// Shared GEMM core: block = 8 waves = 256x64 C region (wave: 32x64).
// Double-buffered TDM pipeline: one barrier per K-step, DMA overlapped.
// ---------------------------------------------------------------------------
static __device__ __forceinline__ void gemm_core(const bf16* __restrict__ Ab,
                                                 const bf16* __restrict__ Wn,
                                                 bf16* __restrict__ ldsB,  // 2*SLAB
                                                 int m0, int lane, int wave,
                                                 v8f c[2][4]) {
  int half = lane >> 4, l16 = lane & 15;
  unsigned lds0 = (unsigned)(uintptr_t)(void*)ldsB;      // LDS byte address = low 32 bits of pointer
  const bf16* arow0 = Ab + (size_t)(m0 + l16) * D_MODEL;
  const bf16* arow1 = Ab + (size_t)(m0 + 16 + l16) * D_MODEL;

  if (wave == 0) tdm_load_slab(Wn, lds0);                // prologue: slab 0
  for (int k0 = 0; k0 < D_MODEL; k0 += 32) {
    int cur = (k0 >> 5) & 1;
    const bf16* bufc = ldsB + cur * SLAB;

    __builtin_amdgcn_s_wait_tensorcnt(0);   // slab k complete (issuing wave)
    __syncthreads();                        // data visible; other buf's readers done
    if (wave == 0 && k0 + 32 < D_MODEL)     // overlap: DMA slab k+1 into other buf
      tdm_load_slab(Wn + k0 + 32, lds0 + (cur ? 0u : (unsigned)(SLAB * sizeof(bf16))));

    v16bf a0 = load_afrag(arow0 + k0, half);
    v16bf a1 = load_afrag(arow1 + k0, half);
    v16bf b[4];
#pragma unroll
    for (int ni = 0; ni < 4; ++ni)
      b[ni] = load_bfrag(bufc + (size_t)(ni * 16 + l16) * 32, half);

#pragma unroll
    for (int ni = 0; ni < 4; ++ni) {
      c[0][ni] = wmma_bf16(a0, b[ni], c[0][ni]);
      c[1][ni] = wmma_bf16(a1, b[ni], c[1][ni]);
    }
  }
}

// ---------------------------------------------------------------------------
// Kernel 2: fused QKV projection. Q,K -> [BH,S,64]; V -> transposed [BH,64,S].
// ---------------------------------------------------------------------------
__global__ void qkv_proj_kernel(const bf16* __restrict__ xb,
                                const bf16* __restrict__ Wt3,
                                const float* __restrict__ bq, const float* __restrict__ bk,
                                const float* __restrict__ bv,
                                bf16* __restrict__ Qb, bf16* __restrict__ Kb,
                                bf16* __restrict__ Vt) {
  __shared__ __align__(16) bf16 ldsB[2 * SLAB];
  int lane = threadIdx.x & 31, wave = threadIdx.x >> 5;
  int id = blockIdx.x;                 // 0 .. 767
  int which = id >> 8;                 // 0=Q 1=K 2=V
  int t = id & 255;
  int m0 = (t >> 4) * 256 + wave * 32;
  int n0 = (t & 15) * 64;
  int half = lane >> 4, l16 = lane & 15;

  v8f c[2][4] = {};
  gemm_core(xb, Wt3 + (size_t)which * D_MODEL * D_MODEL + (size_t)n0 * D_MODEL,
            ldsB, m0, lane, wave, c);

  const float* bias = (which == 0) ? bq : ((which == 1) ? bk : bv);
#pragma unroll
  for (int ni = 0; ni < 4; ++ni) {
    int n = n0 + ni * 16 + l16;
    float bn = bias[n];
    int h = n >> 6, d = n & 63;
#pragma unroll
    for (int mi = 0; mi < 2; ++mi) {
#pragma unroll
      for (int e = 0; e < 8; ++e) {
        int m = m0 + mi * 16 + e + half * 8;       // C row = e + half*8
        int bb = m >> 11, s = m & (SEQ - 1);
        bf16 v = (bf16)(c[mi][ni][e] + bn);
        size_t bh = (size_t)(bb * HEADS + h);
        if (which == 0)      Qb[(bh * SEQ + s) * DK + d] = v;
        else if (which == 1) Kb[(bh * SEQ + s) * DK + d] = v;
        else                 Vt[(bh * DK + d) * SEQ + s] = v;
      }
    }
  }
}

// ---------------------------------------------------------------------------
// Kernel 3: flash-style attention. One wave per (batch*head, 16-query tile).
// 32 keys/iter: 4 WMMAs QK^T, shfl-based online softmax, LDS P-reshape,
// 4 WMMAs P.V.
// ---------------------------------------------------------------------------
__global__ void attn_kernel(const bf16* __restrict__ Qb, const bf16* __restrict__ Kb,
                            const bf16* __restrict__ Vt, bf16* __restrict__ aout) {
  __shared__ __align__(16) bf16 lds[8][16 * 32];
  int lane = threadIdx.x & 31, wave = threadIdx.x >> 5;
  int tile = blockIdx.x * 8 + wave;
  int bh = tile >> 7, qt = tile & 127;
  int q0 = qt * 16;
  int half = lane >> 4, l16 = lane & 15;
  int b = bh >> 4, h = bh & 15;

  const bf16* Qh = Qb + (size_t)bh * SEQ * DK;
  const bf16* Kh = Kb + (size_t)bh * SEQ * DK;
  const bf16* Vh = Vt + (size_t)bh * DK * SEQ;
  bf16* ldsw = &lds[wave][0];

  v16bf qa[2];
  {
    const bf16* qrow = Qh + (size_t)(q0 + l16) * DK;
#pragma unroll
    for (int ks = 0; ks < 2; ++ks) qa[ks] = load_afrag(qrow + ks * 32, half);
  }

  v8f o0 = {}, o1 = {}, o2 = {}, o3 = {};
  float m[8], l[8];
#pragma unroll
  for (int e = 0; e < 8; ++e) { m[e] = -1e30f; l[e] = 0.f; }

  for (int kb = 0; kb < SEQ / 32; ++kb) {
    int keyBase = kb * 32;

    v8f s[2];
#pragma unroll
    for (int ht = 0; ht < 2; ++ht) {
      v8f sc = {};
      const bf16* krow = Kh + (size_t)(keyBase + ht * 16 + l16) * DK;
#pragma unroll
      for (int ks = 0; ks < 2; ++ks) {
        v16bf kf = load_bfrag(krow + ks * 32, half);
        sc = wmma_bf16(qa[ks], kf, sc);
      }
      s[ht] = sc;
    }
#pragma unroll
    for (int e = 0; e < 8; ++e) { s[0][e] *= 0.125f; s[1][e] *= 0.125f; }

    float rmax[8];
#pragma unroll
    for (int e = 0; e < 8; ++e) rmax[e] = fmaxf(s[0][e], s[1][e]);
#pragma unroll
    for (int off = 8; off >= 1; off >>= 1)
#pragma unroll
      for (int e = 0; e < 8; ++e) rmax[e] = fmaxf(rmax[e], __shfl_xor(rmax[e], off, 32));

    float alpha[8], rsum[8];
#pragma unroll
    for (int e = 0; e < 8; ++e) {
      float mn = fmaxf(m[e], rmax[e]);
      alpha[e] = __expf(m[e] - mn);
      m[e] = mn;
      float p0 = __expf(s[0][e] - mn);
      float p1 = __expf(s[1][e] - mn);
      s[0][e] = p0; s[1][e] = p1;
      rsum[e] = p0 + p1;
    }
#pragma unroll
    for (int off = 8; off >= 1; off >>= 1)
#pragma unroll
      for (int e = 0; e < 8; ++e) rsum[e] += __shfl_xor(rsum[e], off, 32);
#pragma unroll
    for (int e = 0; e < 8; ++e) {
      l[e] = l[e] * alpha[e] + rsum[e];
      o0[e] *= alpha[e]; o1[e] *= alpha[e]; o2[e] *= alpha[e]; o3[e] *= alpha[e];
    }

    // reshape P (C layout f32) -> A layout bf16 through per-wave LDS
#pragma unroll
    for (int e = 0; e < 8; ++e) {
      int row = e + half * 8;
      ldsw[row * 32 + l16]      = (bf16)s[0][e];
      ldsw[row * 32 + 16 + l16] = (bf16)s[1][e];
    }
    asm volatile("s_wait_dscnt 0" ::: "memory");
    v16bf pf = load_afrag(ldsw + (size_t)l16 * 32, half);

#pragma unroll
    for (int c4 = 0; c4 < 4; ++c4) {
      const bf16* vrow = Vh + (size_t)(c4 * 16 + l16) * SEQ + keyBase;
      v16bf vf = load_bfrag(vrow, half);
      v8f* op = (c4 == 0) ? &o0 : (c4 == 1) ? &o1 : (c4 == 2) ? &o2 : &o3;
      *op = wmma_bf16(pf, vf, *op);
    }
    asm volatile("s_wait_dscnt 0" ::: "memory");
  }

#pragma unroll
  for (int e = 0; e < 8; ++e) {
    float inv = 1.0f / l[e];
    int srow = q0 + e + half * 8;
    bf16* orow = aout + ((size_t)(b * SEQ + srow)) * D_MODEL + h * DK;
    orow[0 * 16 + l16] = (bf16)(o0[e] * inv);
    orow[1 * 16 + l16] = (bf16)(o1[e] * inv);
    orow[2 * 16 + l16] = (bf16)(o2[e] * inv);
    orow[3 * 16 + l16] = (bf16)(o3[e] * inv);
  }
}

// ---------------------------------------------------------------------------
// Kernel 4: output projection (bf16 x bf16 -> f32) + bias, TDM-staged B.
// ---------------------------------------------------------------------------
__global__ void oproj_kernel(const bf16* __restrict__ A, const bf16* __restrict__ Wot,
                             const float* __restrict__ bo, float* __restrict__ out) {
  __shared__ __align__(16) bf16 ldsB[2 * SLAB];
  int lane = threadIdx.x & 31, wave = threadIdx.x >> 5;
  int id = blockIdx.x;                  // 0 .. 255
  int m0 = (id >> 4) * 256 + wave * 32;
  int n0 = (id & 15) * 64;
  int half = lane >> 4, l16 = lane & 15;

  v8f c[2][4] = {};
  gemm_core(A, Wot + (size_t)n0 * D_MODEL, ldsB, m0, lane, wave, c);

#pragma unroll
  for (int ni = 0; ni < 4; ++ni) {
    int n = n0 + ni * 16 + l16;
    float bn = bo[n];
#pragma unroll
    for (int mi = 0; mi < 2; ++mi)
#pragma unroll
      for (int e = 0; e < 8; ++e)
        out[(size_t)(m0 + mi * 16 + e + half * 8) * D_MODEL + n] = c[mi][ni][e] + bn;
  }
}

// ---------------------------------------------------------------------------
// Launch.  Workspace (48 MiB):
//   [0,8)   MiB : 4x bf16 transposed weights (q,k,v,o)
//   [8,16)  MiB : x cast to bf16 [4096,1024]
//   [16,24) MiB : Q  bf16 [BH,S,64]
//   [24,32) MiB : K  bf16 [BH,S,64]
//   [32,40) MiB : V^T bf16 [BH,64,S]
//   [40,48) MiB : attention output bf16 [B,S,D]
// ---------------------------------------------------------------------------
extern "C" void kernel_launch(void* const* d_in, const int* in_sizes, int n_in,
                              void* d_out, int out_size, void* d_ws, size_t ws_size,
                              hipStream_t stream) {
  (void)in_sizes; (void)n_in; (void)out_size; (void)ws_size;
  const float* x  = (const float*)d_in[0];
  const float* Wq = (const float*)d_in[1];
  const float* bq = (const float*)d_in[2];
  const float* Wk = (const float*)d_in[3];
  const float* bk = (const float*)d_in[4];
  const float* Wv = (const float*)d_in[5];
  const float* bv = (const float*)d_in[6];
  const float* Wo = (const float*)d_in[7];
  const float* bo = (const float*)d_in[8];
  float* out = (float*)d_out;

  char* ws = (char*)d_ws;
  const size_t WT = (size_t)D_MODEL * D_MODEL * sizeof(bf16);        // 2 MiB
  const size_t XS = (size_t)BATCH * SEQ * D_MODEL * sizeof(bf16);    // 8 MiB
  const size_t QS = (size_t)BATCH * HEADS * SEQ * DK * sizeof(bf16); // 8 MiB
  bf16* Wt   = (bf16*)(ws);
  bf16* xb   = (bf16*)(ws + 4 * WT);
  bf16* Qb   = (bf16*)(ws + 4 * WT + XS);
  bf16* Kb   = (bf16*)(ws + 4 * WT + XS + QS);
  bf16* Vt   = (bf16*)(ws + 4 * WT + XS + 2 * QS);
  bf16* attn = (bf16*)(ws + 4 * WT + XS + 3 * QS);

  xcast_kernel<<<dim3((BATCH * SEQ * D_MODEL) / 256), 256, 0, stream>>>(x, xb);
  wtrans_kernel<<<dim3((D_MODEL * D_MODEL) / 256, 4), 256, 0, stream>>>(Wq, Wk, Wv, Wo, Wt);
  qkv_proj_kernel<<<dim3(3 * 256), 256, 0, stream>>>(xb, Wt, bq, bk, bv, Qb, Kb, Vt);
  attn_kernel<<<dim3((32 * 128) / 8), 256, 0, stream>>>(Qb, Kb, Vt, attn);
  oproj_kernel<<<dim3(256), 256, 0, stream>>>(
      attn, Wt + 3 * (size_t)D_MODEL * D_MODEL, bo, out);
}
